// GATModel_30056181137582
// MI455X (gfx1250) — matrix operand order repaired and checked by reading, not compiled, not run on hardware
//
#include <hip/hip_runtime.h>
#include <hip/hip_bf16.h>

// ---------------------------------------------------------------------------
// GAT (3 layers) for MI455X / gfx1250.
// GEMMs via v_wmma_f32_16x16x32_bf16; weights pre-packed to WMMA B-fragment
// order so every lane loads its fragment as 2x b128. Edge softmax/aggregation
// via L2-resident float atomics (working set ~80MB fits the 192MB L2).
// All launches on `stream`; scratch only in d_ws.
// ---------------------------------------------------------------------------

typedef __attribute__((ext_vector_type(16))) __bf16   v16bf;
typedef __attribute__((ext_vector_type(8)))  float    v8f;
typedef __attribute__((ext_vector_type(4)))  unsigned u32x4;
typedef __attribute__((ext_vector_type(4)))  float    f32x4;

union BF16x16 { u32x4 q[2]; v16bf bf; };   // 32B fragment = 2x 16B loads

__device__ __forceinline__ __bf16 f2bf(float f) {
    unsigned u = __float_as_uint(f);
    u += 0x7FFFu + ((u >> 16) & 1u);           // round-to-nearest-even
    unsigned short h = (unsigned short)(u >> 16);
    return __builtin_bit_cast(__bf16, h);
}

// order-preserving float<->uint encoding (atomicMax on floats incl. negatives)
__device__ __forceinline__ unsigned f2ord(float f) {
    unsigned u = __float_as_uint(f);
    return (u & 0x80000000u) ? ~u : (u | 0x80000000u);
}
__device__ __forceinline__ float ord2f(unsigned u) {
    return (u & 0x80000000u) ? __uint_as_float(u & 0x7FFFFFFFu)
                             : __uint_as_float(~u);
}

// --------------------------- cast f32 -> bf16 ------------------------------
__global__ void k_cast_bf16(const float* __restrict__ in, __bf16* __restrict__ out,
                            long n) {
    long i = (long)blockIdx.x * blockDim.x + threadIdx.x;
    if (i < n) out[i] = f2bf(in[i]);
}

// ----------------- pack W [K,M] f32 -> WMMA B-fragment order ---------------
// Bp flat index: (((kt*(M/16) + ct)*16 + col)*2 + half)*16 + elem
//   maps element W[(kt*32 + half*16 + elem)*M + ct*16 + col].
// (B 32x16 fragment, contiguous k order, same as the A loads: lanes 0-15 hold K=kb..kb+15, lanes 16-31
//  hold K=kb+16..kb+31, elements consecutive in K.)
__global__ void k_pack_B(const float* __restrict__ W, __bf16* __restrict__ Bp,
                         int K, int M) {
    long i = (long)blockIdx.x * blockDim.x + threadIdx.x;
    if (i >= (long)K * M) return;
    int elem = (int)(i & 15);
    int half = (int)((i >> 4) & 1);
    int col  = (int)((i >> 5) & 15);
    long rest = i >> 9;
    int ctiles = M >> 4;
    int ct = (int)(rest % ctiles);
    int kt = (int)(rest / ctiles);
    int k = kt * 32 + half * 16 + elem;
    int m = ct * 16 + col;
    Bp[i] = f2bf(W[(size_t)k * M + m]);
}

// --------------------------- WMMA GEMM -------------------------------------
// A: [N,K] bf16 row-major, Bp: packed fragments, C: [N,M] f32.
// One wave per 16x16 output tile; 8 waves per block.
__global__ void k_wmma_gemm(const __bf16* __restrict__ A,
                            const __bf16* __restrict__ Bp,
                            float* __restrict__ C,
                            int N, int K, int M) {
    const int lane     = threadIdx.x & 31;
    const int waveId   = blockIdx.x * (blockDim.x >> 5) + (threadIdx.x >> 5);
    const int colTiles = M >> 4;
    const int rowTile  = waveId / colTiles;
    const int colTile  = waveId - rowTile * colTiles;
    const int row0     = rowTile * 16;
    if (row0 >= N) return;                       // whole-wave uniform exit

    const bool hi = lane >= 16;
    const int  rc = lane & 15;
    const int  r  = row0 + rc;
    const int  rl = r < N ? r : N - 1;           // clamp loads; EXEC stays full
    const __bf16* arow = A + (size_t)rl * K;
    const int  col = colTile * 16 + rc;
    const int  KT  = K >> 5;

    v8f acc = {};
    for (int kt = 0; kt < KT; ++kt) {
        const int kb = kt << 5;
        BF16x16 af, bf;
        // A fragment (16-bit A 16x32 layout): two contiguous 16B chunks
        af.q[0] = *(const u32x4*)(arow + kb + (hi ? 16 : 0));
        af.q[1] = *(const u32x4*)(arow + kb + (hi ? 24 : 8));
        // B fragment: 32B contiguous in packed buffer
        const u32x4* pb = (const u32x4*)
            (Bp + ((((size_t)kt * colTiles + colTile) * 16 + rc) * 2 + (hi ? 1 : 0)) * 16);
        bf.q[0] = pb[0];
        bf.q[1] = pb[1];
        acc = __builtin_amdgcn_wmma_f32_16x16x32_bf16(
                  false, af.bf, false, bf.bf, (short)0, acc, false, false);
    }

    if (row0 + 16 <= N) {                        // full tile: branch-free stores
#pragma unroll
        for (int j = 0; j < 8; ++j)
            C[(size_t)(row0 + j + (hi ? 8 : 0)) * M + col] = acc[j];
    } else {
#pragma unroll
        for (int j = 0; j < 8; ++j) {
            int rr = row0 + j + (hi ? 8 : 0);
            if (rr < N) C[(size_t)rr * M + col] = acc[j];
        }
    }
}

// ------------------- per-(node,head) attention coefficients ----------------
__global__ void k_attn(const float* __restrict__ xl,
                       const float* __restrict__ a_s, const float* __restrict__ a_d,
                       float* __restrict__ als, float* __restrict__ ald,
                       int N, int H, int C) {
    int i = blockIdx.x * blockDim.x + threadIdx.x;
    if (i >= N * H) return;
    int n = i / H, h = i % H;
    const float* v = xl + (size_t)n * H * C + (size_t)h * C;
    float ss = 0.f, sd = 0.f;
    for (int c = 0; c < C; ++c) {
        ss += v[c] * a_s[h * C + c];
        sd += v[c] * a_d[h * C + c];
    }
    als[i] = ss; ald[i] = sd;
}

// ---------------------------- init buffers ---------------------------------
__global__ void k_init(unsigned* __restrict__ m_enc, float* __restrict__ ssum,
                       float* __restrict__ acc, long NH, long NF) {
    long i = (long)blockIdx.x * blockDim.x + threadIdx.x;
    if (i < NH) { m_enc[i] = f2ord(-3.0e38f); ssum[i] = 0.f; }
    if (i < NF) acc[i] = 0.f;
}

// ------------------------- edge pass A: segment max ------------------------
__global__ void k_edge_max(const int* __restrict__ src, const int* __restrict__ dst,
                           int E, int Etot,
                           const float* __restrict__ als, const float* __restrict__ ald,
                           unsigned* __restrict__ m_enc, int H) {
    long t = (long)blockIdx.x * blockDim.x + threadIdx.x;
    if (t >= (long)Etot * H) return;
    int e = (int)(t / H), h = (int)(t % H);
    int s = e < E ? src[e] : e - E;
    int d = e < E ? dst[e] : e - E;
    float v = als[(size_t)s * H + h] + ald[(size_t)d * H + h];
    v = v > 0.f ? v : 0.2f * v;                // leaky relu
    atomicMax(&m_enc[(size_t)d * H + h], f2ord(v));
}

// --------------------- edge pass B: exp + segment sum ----------------------
__global__ void k_edge_exp(const int* __restrict__ src, const int* __restrict__ dst,
                           int E, int Etot,
                           const float* __restrict__ als, const float* __restrict__ ald,
                           const unsigned* __restrict__ m_enc,
                           float* __restrict__ ex, float* __restrict__ ssum, int H) {
    long t = (long)blockIdx.x * blockDim.x + threadIdx.x;
    if (t >= (long)Etot * H) return;
    int e = (int)(t / H), h = (int)(t % H);
    int s = e < E ? src[e] : e - E;
    int d = e < E ? dst[e] : e - E;
    float v = als[(size_t)s * H + h] + ald[(size_t)d * H + h];
    v = v > 0.f ? v : 0.2f * v;
    float m = ord2f(m_enc[(size_t)d * H + h]);
    float x = __expf(v - m);
    ex[(size_t)e * H + h] = x;
    atomicAdd(&ssum[(size_t)d * H + h], x);
}

// ----------------- edge pass C: weighted scatter aggregation ---------------
// one wave per edge; each lane owns 4 consecutive features (b128 gather,
// 4x global_atomic_add_f32 into the L2-resident accumulator)
__global__ void k_edge_aggr(const int* __restrict__ src, const int* __restrict__ dst,
                            int E, int Etot,
                            const float* __restrict__ xl,
                            const float* __restrict__ ex, const float* __restrict__ ssum,
                            float* __restrict__ acc, int H, int cshift, int F) {
    int e = blockIdx.x * (blockDim.x >> 5) + (threadIdx.x >> 5);
    if (e >= Etot) return;
    int lane = threadIdx.x & 31;
    int s = e < E ? src[e] : e - E;
    int d = e < E ? dst[e] : e - E;
    int f0 = lane << 2;
    if (f0 < F) {
        int h = f0 >> cshift;
        float alpha = ex[(size_t)e * H + h] / ssum[(size_t)d * H + h];
        f32x4 xv = *(const f32x4*)(xl + (size_t)s * F + f0);
        float* ap = acc + (size_t)d * F + f0;
        atomicAdd(ap + 0, xv[0] * alpha);
        atomicAdd(ap + 1, xv[1] * alpha);
        atomicAdd(ap + 2, xv[2] * alpha);
        atomicAdd(ap + 3, xv[3] * alpha);
    }
}

// ------------------ bias + ELU + recast to bf16 (layers 0/1) ---------------
__global__ void k_bias_elu_cast(const float* __restrict__ acc,
                                const float* __restrict__ bias,
                                __bf16* __restrict__ hbf, int N, int F) {
    long i = (long)blockIdx.x * blockDim.x + threadIdx.x;
    if (i >= (long)N * F) return;
    int f = (int)(i % F);
    float v = acc[i] + bias[f];
    v = v > 0.f ? v : (__expf(v) - 1.f);       // ELU
    hbf[i] = f2bf(v);
}

// ------------------------ pooling (layer 2 epilogue) -----------------------
__global__ void k_zero(float* __restrict__ p, long n) {
    long i = (long)blockIdx.x * blockDim.x + threadIdx.x;
    if (i < n) p[i] = 0.f;
}

__global__ void k_pool(const float* __restrict__ acc, const float* __restrict__ b2,
                       const int* __restrict__ batch,
                       float* __restrict__ sums, float* __restrict__ cnts,
                       int N, int OUTF) {
    long i = (long)blockIdx.x * blockDim.x + threadIdx.x;
    if (i >= (long)N * OUTF) return;
    int n = (int)(i / OUTF), c = (int)(i % OUTF);
    float v = acc[i] + b2[c];
    int g = batch[n];
    atomicAdd(&sums[(size_t)g * OUTF + c], v);
    if (c == 0) atomicAdd(&cnts[g], 1.0f);
}

__global__ void k_final(const float* __restrict__ sums, const float* __restrict__ cnts,
                        float* __restrict__ out, int G, int OUTF) {
    int i = blockIdx.x * blockDim.x + threadIdx.x;
    if (i >= G * OUTF) return;
    out[i] = sums[i] / fmaxf(cnts[i / OUTF], 1.0f);
}

// ---------------------------------------------------------------------------
static inline unsigned nblk(long n, int b) { return (unsigned)((n + b - 1) / b); }

extern "C" void kernel_launch(void* const* d_in, const int* in_sizes, int n_in,
                              void* d_out, int out_size, void* d_ws, size_t ws_size,
                              hipStream_t stream) {
    const float* x    = (const float*)d_in[0];
    const int*   eidx = (const int*)d_in[1];
    const int*   batch= (const int*)d_in[2];
    const float* W[3]  = {(const float*)d_in[3], (const float*)d_in[7], (const float*)d_in[11]};
    const float* AS[3] = {(const float*)d_in[4], (const float*)d_in[8], (const float*)d_in[12]};
    const float* AD[3] = {(const float*)d_in[5], (const float*)d_in[9], (const float*)d_in[13]};
    const float* BI[3] = {(const float*)d_in[6], (const float*)d_in[10],(const float*)d_in[14]};

    const int N    = in_sizes[2];          // 50000
    const int E    = in_sizes[1] / 2;      // 1.6M
    const int Etot = E + N;                // + self loops
    const int FIN  = in_sizes[0] / N;      // 128
    const int FH   = 128;                  // hidden = H*C for layers 0/1
    const int OUTF = 16;
    const int G    = 64;

    const int* src = eidx;
    const int* dst = eidx + E;

    // ---- workspace carve-up (256B aligned); total ~80 MB, fits L2 ----
    char* base = (char*)d_ws;
    size_t off = 0;
    auto carve = [&](size_t bytes) -> char* {
        char* p = base + off;
        off = (off + bytes + 255) & ~(size_t)255;
        return p;
    };
    __bf16*   hbf   = (__bf16*)  carve((size_t)N * FH * sizeof(__bf16));
    __bf16*   wbf   = (__bf16*)  carve((size_t)FH * FH * sizeof(__bf16));
    float*    xl    = (float*)   carve((size_t)N * FH * sizeof(float));
    float*    acc   = (float*)   carve((size_t)N * FH * sizeof(float));
    float*    als   = (float*)   carve((size_t)N * 2 * sizeof(float));
    float*    ald   = (float*)   carve((size_t)N * 2 * sizeof(float));
    unsigned* menc  = (unsigned*)carve((size_t)N * 2 * sizeof(unsigned));
    float*    ssum  = (float*)   carve((size_t)N * 2 * sizeof(float));
    float*    ex    = (float*)   carve((size_t)Etot * 2 * sizeof(float));
    float*    psum  = (float*)   carve((size_t)G * OUTF * sizeof(float));
    float*    pcnt  = (float*)   carve((size_t)G * sizeof(float));

    // cast input activations to bf16 once
    k_cast_bf16<<<nblk((long)N * FIN, 256), 256, 0, stream>>>(x, hbf, (long)N * FIN);

    const int Kdim[3]   = {FIN, FH, FH};
    const int Mdim[3]   = {FH,  FH, OUTF};
    const int Hh[3]     = {2, 2, 1};
    const int Cc[3]     = {64, 64, OUTF};
    const int Csh[3]    = {6, 6, 4};       // log2(C)

    for (int l = 0; l < 3; ++l) {
        const int Kl = Kdim[l], Ml = Mdim[l], H = Hh[l], C = Cc[l];
        // weights -> packed bf16 B-fragments
        k_pack_B<<<nblk((long)Kl * Ml, 256), 256, 0, stream>>>(W[l], wbf, Kl, Ml);
        // xl = h @ W  (WMMA bf16, f32 accumulate); 8 waves/block, 1 tile/wave
        {
            long totalWaves = (long)((N + 15) / 16) * (Ml / 16);
            k_wmma_gemm<<<nblk(totalWaves, 8), 256, 0, stream>>>(hbf, wbf, xl, N, Kl, Ml);
        }
        // attention coefficients per (node, head)
        k_attn<<<nblk((long)N * H, 256), 256, 0, stream>>>(xl, AS[l], AD[l], als, ald, N, H, C);
        // init segment max / sum / accumulator
        {
            long NH = (long)N * H, NF = (long)N * H * C;
            k_init<<<nblk(NF, 256), 256, 0, stream>>>(menc, ssum, acc, NH, NF);
        }
        // scatter-softmax (3 passes) + aggregation
        k_edge_max<<<nblk((long)Etot * H, 256), 256, 0, stream>>>(src, dst, E, Etot, als, ald, menc, H);
        k_edge_exp<<<nblk((long)Etot * H, 256), 256, 0, stream>>>(src, dst, E, Etot, als, ald, menc, ex, ssum, H);
        k_edge_aggr<<<nblk(Etot, 8), 256, 0, stream>>>(src, dst, E, Etot, xl, ex, ssum, acc, H, Csh[l], H * C);

        if (l < 2) {
            // h = elu(acc + bias) -> bf16 for next layer's GEMM
            k_bias_elu_cast<<<nblk((long)N * H * C, 256), 256, 0, stream>>>(acc, BI[l], hbf, N, H * C);
        } else {
            // global mean pool over graphs (psum & pcnt contiguous after carve)
            k_zero<<<nblk((long)G * OUTF + G, 256), 256, 0, stream>>>(psum, (long)G * OUTF + G);
            k_pool<<<nblk((long)N * OUTF, 256), 256, 0, stream>>>(acc, BI[l], batch, psum, pcnt, N, OUTF);
            k_final<<<nblk(G * OUTF, 256), 256, 0, stream>>>(psum, pcnt, (float*)d_out, G, OUTF);
        }
    }
}
